// RMTBlock_12807592476918
// MI455X (gfx1250) — matrix operand order repaired and checked by reading, not compiled, not run on hardware
//
#include <hip/hip_runtime.h>
#include <hip/hip_bf16.h>

// ---------------------------------------------------------------------------
// RMT transformer block on gfx1250 (MI455X): f16 WMMA + f32 accum.
//   B=2, N=4096, C=256, heads=4, d=64, FFN=1024.
// cvt->f16, QKV GEMM (scatter q/k/vT), TDM-fed flash-attention (wave per
// 32-query tile, DPP softmax reductions), out-proj GEMM, fc1 (ReLU),
// fc2 (+bias+residual) -> f32 out.
// ---------------------------------------------------------------------------

typedef _Float16 v16h __attribute__((ext_vector_type(16)));
typedef _Float16 h2v  __attribute__((ext_vector_type(2)));
typedef float    v8f  __attribute__((ext_vector_type(8)));
typedef unsigned int u32x4 __attribute__((ext_vector_type(4)));
typedef int      i32x4 __attribute__((ext_vector_type(4)));
typedef int      i32x8 __attribute__((ext_vector_type(8)));

#define B_   2
#define N_   4096
#define C_   256
#define H_   4
#define D_   64
#define M_   (B_ * N_)   /* 8192 rows in every GEMM */
#define FF_  1024

// ---------------------------------------------------------------------------
// WMMA helpers (ISA 7.12.2 layouts, wave32)
// ---------------------------------------------------------------------------
__device__ __forceinline__ v8f wmma16(v16h a, v16h b, v8f c) {
  return __builtin_amdgcn_wmma_f32_16x16x32_f16(false, a, false, b, (short)0, c,
                                                false, false);
}

// A matrix 16x32 (MxK) f16: lane = row (mod16); element e <-> K = hf*16+e
// (same contiguous order as load_b16).
__device__ __forceinline__ v16h load_a16(const _Float16* base, int ld, int m0,
                                         int k0) {
  const int l = threadIdx.x & 31;
  const int row = l & 15, hf = l >> 4;
  return *(const v16h*)(base + (size_t)(m0 + row) * ld + k0 + hf * 16);
}

// B matrix 32x16 (KxN) f16: lane = col (mod16); lanes 0-15: K=0..15,
// lanes 16-31: K=16..31 (element e <-> K = hf*16+e): one contiguous 32B read.
__device__ __forceinline__ v16h load_b16(const _Float16* base, int ld, int n0,
                                         int k0) {
  const int l = threadIdx.x & 31;
  const int col = l & 15, hf = l >> 4;
  return *(const v16h*)(base + (size_t)(n0 + col) * ld + k0 + hf * 16);
}

// ---------------------------------------------------------------------------
// DPP16 row reductions (16-lane rows == C/D-layout softmax rows).
// quad_perm xor1 (0xB1), xor2 (0x4E), then ROW_HALF_MIRROR (0x141) == xor4 and
// ROW_MIRROR (0x140) == xor8 once sub-groups are uniform. Pure VALU, no LDS.
// ---------------------------------------------------------------------------
template <int CTRL>
__device__ __forceinline__ float dppf(float x) {
  return __int_as_float(__builtin_amdgcn_update_dpp(
      0, __float_as_int(x), CTRL, 0xF, 0xF, true));
}
__device__ __forceinline__ float rowmax16(float x) {
  x = fmaxf(x, dppf<0xB1>(x));
  x = fmaxf(x, dppf<0x4E>(x));
  x = fmaxf(x, dppf<0x141>(x));
  x = fmaxf(x, dppf<0x140>(x));
  return x;
}
__device__ __forceinline__ float rowsum16(float x) {
  x += dppf<0xB1>(x);
  x += dppf<0x4E>(x);
  x += dppf<0x141>(x);
  x += dppf<0x140>(x);
  return x;
}

// ---------------------------------------------------------------------------
// Tensor Data Mover: 2D tile load (CDNA5 ISA D# layout).
// dims/strides in 2-byte elements (data_size=1). Tracked by TENSORcnt.
// ---------------------------------------------------------------------------
__device__ __forceinline__ void tdm_load_2d(unsigned lds_byte_addr,
                                            const void* gaddr, unsigned tile_d0,
                                            unsigned tile_d1,
                                            unsigned tensor_d0,
                                            unsigned tensor_d1,
                                            unsigned stride0_elems) {
  unsigned long long ga = (unsigned long long)gaddr;
  u32x4 g0;
  g0[0] = 1u;                                     // count=1, user mode
  g0[1] = lds_byte_addr;                          // lds_addr
  g0[2] = (unsigned)(ga & 0xffffffffu);           // global_addr[31:0]
  g0[3] = (unsigned)((ga >> 32) & 0x1ffffffu) | (2u << 30);  // [120:64],type=2
  i32x8 g1;
  g1[0] = (int)(1u << 16);                        // data_size=1 (2B), mask=0
  g1[1] = (int)((tensor_d0 & 0xffffu) << 16);     // tensor_dim0[15:0]
  g1[2] = (int)(((tensor_d0 >> 16) & 0xffffu) | ((tensor_d1 & 0xffffu) << 16));
  g1[3] = (int)(((tensor_d1 >> 16) & 0xffffu) | ((tile_d0 & 0xffffu) << 16));
  g1[4] = (int)(tile_d1 & 0xffffu);               // tile_dim1, tile_dim2=0
  g1[5] = (int)stride0_elems;                     // tensor_dim0_stride[31:0]
  g1[6] = 0;
  g1[7] = 0;
  i32x4 z4 = {0, 0, 0, 0};
#if defined(__clang_major__) && __clang_major__ >= 23
  i32x8 z8 = {0, 0, 0, 0, 0, 0, 0, 0};
  __builtin_amdgcn_tensor_load_to_lds(g0, g1, z4, z4, z8, 0);
#else
  __builtin_amdgcn_tensor_load_to_lds(g0, g1, z4, z4, 0);
#endif
}

// ---------------------------------------------------------------------------
// f32 -> f16 conversion
// ---------------------------------------------------------------------------
__global__ void cvt_f32_to_f16(const float* __restrict__ in,
                               _Float16* __restrict__ out, int n) {
  int i = blockIdx.x * blockDim.x + threadIdx.x;
  if (i < n) out[i] = (_Float16)in[i];
}

// ---------------------------------------------------------------------------
// Generic WMMA GEMM:  Y[m,n] = sum_k X[m,k] * W[n,k]  (+ bias, + epilogue)
// Block = 128 threads (4 waves). Wave: 32 rows x 64 cols (2 A sets share every
// B fragment -> half the W traffic). K-loop step 32.
// MODE 0: scatter q[bh,N,d] / k[bh,N,d] / vT[bh,d,N]     (QKV proj)
// MODE 1: write f32 (residual keep) + f16 copy           (out proj)
// MODE 2: ReLU -> f16                                    (fc1)
// MODE 3: f32 = acc + bias + res[m,n]                    (fc2 + residual)
// ---------------------------------------------------------------------------
template <int MODE>
__global__ __launch_bounds__(128) void gemm_wmma(
    const _Float16* __restrict__ X, const _Float16* __restrict__ W,
    const float* __restrict__ bias, int K, int Nout, float* __restrict__ outf,
    _Float16* __restrict__ outh, const float* __restrict__ res,
    _Float16* __restrict__ qh, _Float16* __restrict__ kh,
    _Float16* __restrict__ vT) {
  const int wave = threadIdx.x >> 5;
  const int m0 = blockIdx.y * 128 + wave * 32;
  const int n0 = blockIdx.x * 64;
  const int l = threadIdx.x & 31;
  const int col = l & 15, hf = l >> 4;

  v8f z = {};
  v8f acc[2][4];
#pragma unroll
  for (int s = 0; s < 2; ++s)
#pragma unroll
    for (int t = 0; t < 4; ++t) acc[s][t] = z;

  for (int k0 = 0; k0 < K; k0 += 32) {
    v16h a0 = load_a16(X, K, m0, k0);
    v16h a1 = load_a16(X, K, m0 + 16, k0);
#pragma unroll
    for (int t = 0; t < 4; ++t) {
      v16h b = load_b16(W, K, n0 + t * 16, k0);
      acc[0][t] = wmma16(a0, b, acc[0][t]);
      acc[1][t] = wmma16(a1, b, acc[1][t]);
    }
  }

#pragma unroll
  for (int s = 0; s < 2; ++s)
#pragma unroll
    for (int t = 0; t < 4; ++t) {
      const int n = n0 + t * 16 + col;
      const float bv = bias ? bias[n] : 0.0f;
#pragma unroll
      for (int v = 0; v < 8; ++v) {
        const int m = m0 + s * 16 + v + hf * 8;  // C/D layout row
        float val = acc[s][t][v] + bv;
        if (MODE == 0) {
          const int r = n >> 8;  // 0=q 1=k 2=v (uniform per wave)
          const int c = n & 255;
          const int head = c >> 6, dd = c & 63;
          const int b = m >> 12, tok = m & 4095;
          const size_t bh = (size_t)(b * H_ + head);
          if (r == 0)
            qh[(bh * N_ + tok) * D_ + dd] = (_Float16)val;
          else if (r == 1)
            kh[(bh * N_ + tok) * D_ + dd] = (_Float16)val;
          else
            vT[(bh * D_ + dd) * N_ + tok] = (_Float16)val;
        } else if (MODE == 1) {
          outf[(size_t)m * Nout + n] = val;
          outh[(size_t)m * Nout + n] = (_Float16)val;
        } else if (MODE == 2) {
          val = val > 0.0f ? val : 0.0f;
          outh[(size_t)m * Nout + n] = (_Float16)val;
        } else {
          outf[(size_t)m * Nout + n] = val + res[(size_t)m * Nout + n];
        }
      }
    }
}

// ---------------------------------------------------------------------------
// Flash attention: one wave per (b,h, 32-query tile). K/V chunks of 32 keys
// double-buffered in LDS via TENSOR_LOAD_TO_LDS (TENSORcnt pipeline);
// online softmax in f32 with DPP row reductions; P converted C/D -> A layout
// through a 2KB LDS tile. Single-wave workgroup => barriers are S_NOP.
// ---------------------------------------------------------------------------
__global__ __launch_bounds__(32) void flash_attn(
    const _Float16* __restrict__ qh, const _Float16* __restrict__ kh,
    const _Float16* __restrict__ vT, _Float16* __restrict__ ctxh) {
  __shared__ __attribute__((aligned(32))) _Float16 Kb[2][32 * 64];  // [key][d]
  __shared__ __attribute__((aligned(32))) _Float16 Vb[2][64 * 32];  // [d][key]
  __shared__ __attribute__((aligned(32))) _Float16 P[32 * 32];      // [q][key]

  const int bh = blockIdx.y;  // 0..B*H-1
  const int b = bh >> 2, h = bh & 3;
  const int m0 = blockIdx.x * 32;
  const _Float16* qp = qh + (size_t)bh * N_ * D_;
  const _Float16* kp = kh + (size_t)bh * N_ * D_;
  const _Float16* vp = vT + (size_t)bh * D_ * N_;
  const int l = threadIdx.x & 31;
  const int col = l & 15, hf = l >> 4;

  // Q fragments fixed for the whole tile: 2 row sets x (d = 2 x K32)
  v16h aq[2][2];
#pragma unroll
  for (int s = 0; s < 2; ++s) {
    aq[s][0] = load_a16(qp, D_, m0 + s * 16, 0);
    aq[s][1] = load_a16(qp, D_, m0 + s * 16, 32);
  }

  v8f z = {};
  v8f acc[2][4];
  float mrun[2][8], lrun[2][8];
#pragma unroll
  for (int s = 0; s < 2; ++s)
#pragma unroll
    for (int t = 0; t < 4; ++t) acc[s][t] = z;
#pragma unroll
  for (int s = 0; s < 2; ++s)
#pragma unroll
    for (int v = 0; v < 8; ++v) {
      mrun[s][v] = -1.0e30f;
      lrun[s][v] = 0.0f;
    }

  auto prefetch = [&](int buf, int key0) {
    // K tile: 32 rows (keys) x 64 d, row stride 64 elems, contiguous dest.
    tdm_load_2d((unsigned)(size_t)&Kb[buf][0], kp + (size_t)key0 * D_, D_, 32,
                D_, N_, D_);
    // V tile: 64 rows (d) x 32 keys, row stride N elems.
    tdm_load_2d((unsigned)(size_t)&Vb[buf][0], vp + key0, 32, D_, N_, D_, N_);
  };

  prefetch(0, 0);
  const int NCHUNK = N_ / 32;
  for (int ch = 0; ch < NCHUNK; ++ch) {
    const int buf = ch & 1;
    if (ch + 1 < NCHUNK) {
      prefetch(buf ^ 1, (ch + 1) * 32);
      __builtin_amdgcn_s_wait_tensorcnt(2);  // current chunk's 2 loads done
    } else {
      __builtin_amdgcn_s_wait_tensorcnt(0);
    }
    asm volatile("" ::: "memory");  // keep LDS reads after the wait

    const _Float16* Kc = &Kb[buf][0];
    const _Float16* Vc = &Vb[buf][0];
    v16h bk[2][2];
#pragma unroll
    for (int nt = 0; nt < 2; ++nt) {
      bk[nt][0] = load_b16(Kc, D_, nt * 16, 0);
      bk[nt][1] = load_b16(Kc, D_, nt * 16, 32);
    }

    __syncthreads();  // S_NOP; protect P vs previous A-fragment reads
    float alpha[2][8];
#pragma unroll
    for (int s = 0; s < 2; ++s) {
      v8f s0 = wmma16(aq[s][0], bk[0][0], z);
      s0 = wmma16(aq[s][1], bk[0][1], s0);
      v8f s1 = wmma16(aq[s][0], bk[1][0], z);
      s1 = wmma16(aq[s][1], bk[1][1], s1);
#pragma unroll
      for (int v = 0; v < 8; ++v) {
        const float a0 = s0[v] * 0.125f + 1.0f;
        const float a1 = s1[v] * 0.125f + 1.0f;
        const float rm = rowmax16(fmaxf(a0, a1));
        const float nm = fmaxf(mrun[s][v], rm);
        alpha[s][v] = __expf(mrun[s][v] - nm);
        mrun[s][v] = nm;
        const float p0 = __expf(a0 - nm);
        const float p1 = __expf(a1 - nm);
        lrun[s][v] = lrun[s][v] * alpha[s][v] + rowsum16(p0 + p1);
        const int row = s * 16 + v + hf * 8;
        P[row * 32 + col] = (_Float16)p0;
        P[row * 32 + 16 + col] = (_Float16)p1;
      }
#pragma unroll
      for (int t = 0; t < 4; ++t)
#pragma unroll
        for (int v = 0; v < 8; ++v) acc[s][t][v] *= alpha[s][v];
    }
    __syncthreads();  // S_NOP; P writes -> A-fragment reads

    v16h ap0 = load_a16(&P[0], 32, 0, 0);
    v16h ap1 = load_a16(&P[16 * 32], 32, 0, 0);
#pragma unroll
    for (int t = 0; t < 4; ++t) {
      v16h bv = load_b16(Vc, 32, t * 16, 0);
      acc[0][t] = wmma16(ap0, bv, acc[0][t]);
      acc[1][t] = wmma16(ap1, bv, acc[1][t]);
    }
  }

  // ctx[b, tok, h*64 + dd] = acc / l
#pragma unroll
  for (int s = 0; s < 2; ++s)
#pragma unroll
    for (int t = 0; t < 4; ++t)
#pragma unroll
      for (int v = 0; v < 8; ++v) {
        const int m = m0 + s * 16 + v + hf * 8;
        const int cc = h * D_ + t * 16 + col;
        ctxh[((size_t)(b * N_ + m)) * C_ + cc] =
            (_Float16)(acc[s][t][v] / lrun[s][v]);
      }
}

// ---------------------------------------------------------------------------
// Host-side launch
// ---------------------------------------------------------------------------
extern "C" void kernel_launch(void* const* d_in, const int* in_sizes, int n_in,
                              void* d_out, int out_size, void* d_ws,
                              size_t ws_size, hipStream_t stream) {
  const float* x    = (const float*)d_in[0];
  const float* inW  = (const float*)d_in[1];
  const float* inB  = (const float*)d_in[2];
  const float* outW = (const float*)d_in[3];
  const float* outB = (const float*)d_in[4];
  const float* fc1W = (const float*)d_in[5];
  const float* fc1B = (const float*)d_in[6];
  const float* fc2W = (const float*)d_in[7];
  const float* fc2B = (const float*)d_in[8];

  char* ws = (char*)d_ws;
  size_t off = 0;
  auto take = [&](size_t bytes) {
    void* p = ws + off;
    off = (off + bytes + 255) & ~(size_t)255;
    return p;
  };
  _Float16* xh    = (_Float16*)take((size_t)M_ * C_ * 2);
  _Float16* wqkvh = (_Float16*)take((size_t)3 * C_ * C_ * 2);
  _Float16* wouth = (_Float16*)take((size_t)C_ * C_ * 2);
  _Float16* wfc1h = (_Float16*)take((size_t)FF_ * C_ * 2);
  _Float16* wfc2h = (_Float16*)take((size_t)C_ * FF_ * 2);
  _Float16* q_h   = (_Float16*)take((size_t)B_ * H_ * N_ * D_ * 2);
  _Float16* k_h   = (_Float16*)take((size_t)B_ * H_ * N_ * D_ * 2);
  _Float16* vT_h  = (_Float16*)take((size_t)B_ * H_ * D_ * N_ * 2);
  _Float16* ctxh  = (_Float16*)take((size_t)M_ * C_ * 2);
  float*    attnf = (float*)take((size_t)M_ * C_ * 4);
  _Float16* attnh = (_Float16*)take((size_t)M_ * C_ * 2);
  _Float16* ffnh  = (_Float16*)take((size_t)M_ * FF_ * 2);

  auto cvt = [&](const float* src, _Float16* dst, int n) {
    cvt_f32_to_f16<<<(n + 255) / 256, 256, 0, stream>>>(src, dst, n);
  };
  cvt(x, xh, M_ * C_);
  cvt(inW, wqkvh, 3 * C_ * C_);
  cvt(outW, wouth, C_ * C_);
  cvt(fc1W, wfc1h, FF_ * C_);
  cvt(fc2W, wfc2h, C_ * FF_);

  gemm_wmma<0><<<dim3(3 * C_ / 64, M_ / 128), 128, 0, stream>>>(
      xh, wqkvh, inB, C_, 3 * C_, nullptr, nullptr, nullptr, q_h, k_h, vT_h);

  flash_attn<<<dim3(N_ / 32, B_ * H_), 32, 0, stream>>>(q_h, k_h, vT_h, ctxh);

  gemm_wmma<1><<<dim3(C_ / 64, M_ / 128), 128, 0, stream>>>(
      ctxh, wouth, outB, C_, C_, attnf, attnh, nullptr, nullptr, nullptr,
      nullptr);

  gemm_wmma<2><<<dim3(FF_ / 64, M_ / 128), 128, 0, stream>>>(
      attnh, wfc1h, fc1B, C_, FF_, nullptr, ffnh, nullptr, nullptr, nullptr,
      nullptr);

  gemm_wmma<3><<<dim3(C_ / 64, M_ / 128), 128, 0, stream>>>(
      ffnh, wfc2h, fc2B, FF_, C_, (float*)d_out, nullptr, attnf, nullptr,
      nullptr, nullptr);
}
